// VisionTransformer_25348896981091
// MI455X (gfx1250) — matrix operand order repaired and checked by reading, not compiled, not run on hardware
//
#include <hip/hip_runtime.h>

// ---------------------------------------------------------------------------
// Problem constants (from reference)
// ---------------------------------------------------------------------------
#define BB   8        // batch
#define NI   4        // images
#define SS   576      // patches per image
#define DD   256      // model dim
#define NH   8        // heads
#define DH   32       // head dim
#define LL   6        // layers
#define KPE  768      // patch-embed K = 3*16*16
#define MLPD 512

typedef __attribute__((ext_vector_type(16))) __bf16 v16bf;
typedef __attribute__((ext_vector_type(8)))  float  v8f;

__device__ inline v8f zero8() {
  v8f z = {0.f, 0.f, 0.f, 0.f, 0.f, 0.f, 0.f, 0.f};
  return z;
}

__device__ inline v8f wmma_bf16(v16bf a, v16bf b, v8f c) {
  return __builtin_amdgcn_wmma_f32_16x16x32_bf16(false, a, false, b, (short)0, c,
                                                 false, false);
}

// A fragment (16x32 bf16). Layout: lanes 0-15 rows M=0..15, K 0..15;
// lanes 16-31 same rows, K 16..31 (contiguous, matches load_b_frag).
template <typename T>
__device__ inline v16bf load_a_frag(const T* A, int lda, int m0, int k0) {
  const int lane = threadIdx.x & 31;
  const T* p = A + (size_t)(m0 + (lane & 15)) * lda + k0 + ((lane >> 4) << 4);
  v16bf a;
#pragma unroll
  for (int e = 0; e < 8; ++e) a[e] = (__bf16)(float)p[e];
#pragma unroll
  for (int e = 0; e < 8; ++e) a[e + 8] = (__bf16)(float)p[e + 8];
  return a;
}

// B fragment (32x16 bf16) loaded from B^T stored row-major [N][K]:
// lane holds column N = n0 + lane%16, contiguous K chunk of 16 (+16 hi half).
template <typename T>
__device__ inline v16bf load_b_frag(const T* Bt, int ldb, int n0, int k0) {
  const int lane = threadIdx.x & 31;
  const T* p = Bt + (size_t)(n0 + (lane & 15)) * ldb + k0 + ((lane >> 4) << 4);
  v16bf b;
#pragma unroll
  for (int e = 0; e < 16; ++e) b[e] = (__bf16)(float)p[e];
  return b;
}

// ---------------------------------------------------------------------------
// One-shot fp32 -> bf16 weight conversion
// ---------------------------------------------------------------------------
__global__ void cvt_f32_bf16_kernel(const float* __restrict__ src,
                                    __bf16* __restrict__ dst, int n) {
  const int i = blockIdx.x * 256 + threadIdx.x;
  if (i < n) dst[i] = (__bf16)src[i];
}

// ---------------------------------------------------------------------------
// Patch embed: pe[i,b,s,d] = im2col(x) @ conv_w[i]^T + conv_b[i,d] + pos[s,d]
// GEMM per image: M=4608 (b,s), K=768, N=256.  Block: 256 thr, tile 128x32.
// ---------------------------------------------------------------------------
__global__ void patch_embed_kernel(const float* __restrict__ x,
                                   const __bf16* __restrict__ cw,
                                   const float* __restrict__ cb,
                                   const float* __restrict__ pos,
                                   __bf16* __restrict__ pe) {
  const int i    = blockIdx.z;
  const int m0   = blockIdx.x * 128;
  const int n0   = blockIdx.y * 32;
  const int tid  = threadIdx.x;
  const int wave = tid >> 5;
  const int lane = tid & 31;

  __shared__ __bf16 At[128][40];  // 128x32 tile, padded

  const __bf16* cwb = cw + (size_t)i * DD * KPE;
  v8f acc0 = zero8(), acc1 = zero8();

  for (int k0 = 0; k0 < KPE; k0 += 32) {
    // cooperative im2col gather into LDS (4096 elems / 256 threads)
#pragma unroll
    for (int t = 0; t < 16; ++t) {
      const int e  = tid + 256 * t;
      const int r  = e >> 5;
      const int kk = e & 31;
      const int g  = m0 + r;          // global row = b*576 + s
      const int b  = g / SS;
      const int s  = g % SS;
      const int py = s / 24, px = s % 24;
      const int k  = k0 + kk;
      const int c  = k >> 8;
      const int rem = k & 255;
      const int dy = rem >> 4, dx = rem & 15;
      At[r][kk] = (__bf16)x[(((size_t)b * (NI * 3) + i * 3 + c) * 384 +
                             (py * 16 + dy)) * 384 + (px * 16 + dx)];
    }
    __syncthreads();
    v16bf a  = load_a_frag(&At[0][0], 40, wave * 16, 0);
    v16bf b0 = load_b_frag(cwb, KPE, n0, k0);
    v16bf b1 = load_b_frag(cwb, KPE, n0 + 16, k0);
    acc0 = wmma_bf16(a, b0, acc0);
    acc1 = wmma_bf16(a, b1, acc1);
    __syncthreads();
  }

  const int colb = lane & 15;
  const int rb   = (lane >> 4) << 3;
  __bf16* peI = pe + (size_t)i * (BB * SS) * DD;
#pragma unroll
  for (int r = 0; r < 8; ++r) {
    const int grow = m0 + wave * 16 + rb + r;
    const int s = grow % SS;
    int col = n0 + colb;
    peI[(size_t)grow * DD + col] =
        (__bf16)(acc0[r] + cb[i * DD + col] + pos[(size_t)s * DD + col]);
    col += 16;
    peI[(size_t)grow * DD + col] =
        (__bf16)(acc1[r] + cb[i * DD + col] + pos[(size_t)s * DD + col]);
  }
}

// ---------------------------------------------------------------------------
// Generic GEMM + bias:  C[M,N] = A[M,K] @ W^T + bias,  W (bf16) row-major [N][K].
// Used for QKV (N=768, also scatters V^T) and out-proj (N=256).
// Block: 256 thr, tile 128x32, wave owns one 16-row slab x both 16-col tiles.
// ---------------------------------------------------------------------------
__global__ void gemm_bias_kernel(const __bf16* __restrict__ A,
                                 const __bf16* __restrict__ Bt,
                                 const float* __restrict__ bias,
                                 __bf16* __restrict__ C,
                                 __bf16* __restrict__ vT,  // nullptr if unused
                                 int K, int N) {
  const int m0   = blockIdx.x * 128 + (threadIdx.x >> 5) * 16;
  const int n0   = blockIdx.y * 32;
  const int lane = threadIdx.x & 31;

  v8f acc0 = zero8(), acc1 = zero8();
  for (int k0 = 0; k0 < K; k0 += 32) {
    // prefetch next K-chunk (global_prefetch_b8; speculative, safe at tail)
    __builtin_prefetch(A + (size_t)(m0 + (lane & 15)) * K + k0 + 32, 0, 1);
    __builtin_prefetch(Bt + (size_t)(n0 + lane) * K + k0 + 32, 0, 1);
    v16bf a  = load_a_frag(A, K, m0, k0);
    v16bf b0 = load_b_frag(Bt, K, n0, k0);
    v16bf b1 = load_b_frag(Bt, K, n0 + 16, k0);
    acc0 = wmma_bf16(a, b0, acc0);
    acc1 = wmma_bf16(a, b1, acc1);
  }

  const int colb = lane & 15;
  const int rb   = (lane >> 4) << 3;
#pragma unroll
  for (int r = 0; r < 8; ++r) {
    const int row = m0 + rb + r;
#pragma unroll
    for (int t = 0; t < 2; ++t) {
      const int col = n0 + t * 16 + colb;
      const float v = (t ? acc1[r] : acc0[r]) + bias[col];
      C[(size_t)row * N + col] = (__bf16)v;
      if (vT != nullptr && col >= 512) {  // V portion -> transposed copy
        const int ib = row / SS, s = row % SS;
        vT[((size_t)ib * DD + (col - 512)) * SS + s] = (__bf16)v;
      }
    }
  }
}

// ---------------------------------------------------------------------------
// Fused attention for one (ib, 16-q-row) slab; loops over the 8 heads.
// qkv[ib][s][0:256]=Q, [256:512]=K, [512:768]=V;  vT[ib][d][s]=V[s][d].
// Writes O slab into obuf (bf16) and head-averaged probs (fp32) into attnL.
// ---------------------------------------------------------------------------
__global__ void attention_kernel(const __bf16* __restrict__ qkv,
                                 const __bf16* __restrict__ vT,
                                 __bf16* __restrict__ obuf,
                                 float* __restrict__ attnL) {
  const int q0   = blockIdx.x * 16;   // 36 q-tiles
  const int ib   = blockIdx.y;        // 32 (image,batch) pairs
  const int tid  = threadIdx.x;
  const int wave = tid >> 5;
  const int lane = tid & 31;

  __shared__ float sc[16][592];       // scores / probs for current head
  __shared__ float avg[16][592];      // head-averaged probs accumulator
  __shared__ float red[8][16][16];    // K-split partial reduction for attn@V

  for (int idx = tid; idx < 16 * SS; idx += 256) avg[idx / SS][idx % SS] = 0.f;
  __syncthreads();

  const __bf16* qbase = qkv + (size_t)ib * SS * (3 * DD);
  const float scale = 0.17677669529663687f;  // 1/sqrt(32)

  for (int h = 0; h < NH; ++h) {
    // ---- scores = Q_h @ K_h^T * scale : M=16, N=576, K=32 (one WMMA each)
    v16bf aq = load_a_frag(qbase + h * DH, 3 * DD, q0, 0);
    for (int nt = wave; nt < 36; nt += 8) {
      v16bf bk = load_b_frag(qbase + DD + h * DH, 3 * DD, nt * 16, 0);
      v8f s = zero8();
      s = wmma_bf16(aq, bk, s);
      const int col = nt * 16 + (lane & 15);
      const int rb  = (lane >> 4) << 3;
#pragma unroll
      for (int r = 0; r < 8; ++r) sc[rb + r][col] = s[r] * scale;
    }
    __syncthreads();

    // ---- softmax per row (16 threads per row; rows live in wave32 halves)
    {
      const int row = tid >> 4, sub = tid & 15;
      float m = -3.4e38f;
      for (int c = sub; c < SS; c += 16) m = fmaxf(m, sc[row][c]);
#pragma unroll
      for (int off = 1; off < 16; off <<= 1) m = fmaxf(m, __shfl_xor(m, off, 32));
      float sum = 0.f;
      for (int c = sub; c < SS; c += 16) {
        const float e = __expf(sc[row][c] - m);
        sc[row][c] = e;
        sum += e;
      }
#pragma unroll
      for (int off = 1; off < 16; off <<= 1) sum += __shfl_xor(sum, off, 32);
      const float inv = 1.f / sum;
      for (int c = sub; c < SS; c += 16) {
        const float p = sc[row][c] * inv;
        sc[row][c] = p;
        avg[row][c] += p * 0.125f;
      }
    }
    __syncthreads();

    // ---- O_h = probs @ V_h : M=16, N=32, K=576 (K split 4 ways per n-tile)
    {
      const int nt = wave >> 2;  // 0..1 (16-col tile of DH)
      const int kc = wave & 3;   // K-chunk owner
      const __bf16* vbase = vT + ((size_t)ib * DD + h * DH + nt * 16) * SS;
      v8f acc = zero8();
      for (int it = kc; it < 18; it += 4) {
        const int k0 = it * 32;
        v16bf a = load_a_frag(&sc[0][0], 592, 0, k0);
        v16bf b = load_b_frag(vbase, SS, 0, k0);
        acc = wmma_bf16(a, b, acc);
      }
      const int col = lane & 15;
      const int rb  = (lane >> 4) << 3;
#pragma unroll
      for (int r = 0; r < 8; ++r) red[wave][rb + r][col] = acc[r];
    }
    __syncthreads();

    // ---- reduce the 4 partials per n-tile, store O slab
    {
      const int r = tid >> 4, c = tid & 15;
#pragma unroll
      for (int nt = 0; nt < 2; ++nt) {
        const float s = red[nt * 4 + 0][r][c] + red[nt * 4 + 1][r][c] +
                        red[nt * 4 + 2][r][c] + red[nt * 4 + 3][r][c];
        obuf[((size_t)ib * SS + q0 + r) * DD + h * DH + nt * 16 + c] =
            (__bf16)s;
      }
    }
    __syncthreads();
  }

  // ---- head-averaged attention weights -> d_out slice (contiguous slab)
  float* dst = attnL + ((size_t)ib * SS + q0) * SS;
  for (int idx = tid; idx < 16 * SS; idx += 256)
    dst[idx] = avg[idx / SS][idx % SS];
}

// ---------------------------------------------------------------------------
// Head: mean over patches, FC(1024->512)+ReLU, FC(512->1)
// ---------------------------------------------------------------------------
__global__ void feat_mean_kernel(const __bf16* __restrict__ pe,
                                 float* __restrict__ feat) {
  const int idx = blockIdx.x * 256 + threadIdx.x;  // 8192 = 8 * 1024
  const int b = idx >> 10;
  const int rem = idx & 1023;
  const int i = rem >> 8;
  const int d = rem & 255;
  float sum = 0.f;
  const __bf16* p = pe + ((size_t)(i * BB + b) * SS) * DD + d;
  for (int s = 0; s < SS; ++s) sum += (float)p[(size_t)s * DD];
  feat[idx] = sum * (1.0f / SS);
}

__global__ void head1_kernel(const float* __restrict__ feat,
                             const float* __restrict__ w1,
                             const float* __restrict__ b1,
                             float* __restrict__ h) {
  const int idx = blockIdx.x * 256 + threadIdx.x;  // 4096 = 8 * 512
  const int b = idx >> 9;
  const int m = idx & 511;
  float sum = b1[m];
  const float* f = feat + (size_t)b * (NI * DD);
  const float* w = w1 + (size_t)m * (NI * DD);
  for (int k = 0; k < NI * DD; ++k) sum += f[k] * w[k];
  h[idx] = sum > 0.f ? sum : 0.f;
}

__global__ void head2_kernel(const float* __restrict__ h,
                             const float* __restrict__ w2,
                             const float* __restrict__ b2,
                             float* __restrict__ out) {
  const int b = threadIdx.x;
  if (b < BB) {
    float sum = b2[0];
    for (int m = 0; m < MLPD; ++m) sum += h[(size_t)b * MLPD + m] * w2[m];
    out[b] = sum;
  }
}

// ---------------------------------------------------------------------------
// Launch
// ---------------------------------------------------------------------------
extern "C" void kernel_launch(void* const* d_in, const int* in_sizes, int n_in,
                              void* d_out, int out_size, void* d_ws,
                              size_t ws_size, hipStream_t stream) {
  (void)in_sizes; (void)n_in; (void)out_size; (void)ws_size;
  const float* x      = (const float*)d_in[0];
  const float* conv_w = (const float*)d_in[1];
  const float* conv_b = (const float*)d_in[2];
  const float* pos    = (const float*)d_in[3];
  const float* inw    = (const float*)d_in[4];
  const float* inb    = (const float*)d_in[5];
  const float* outw   = (const float*)d_in[6];
  const float* outb   = (const float*)d_in[7];
  const float* w1     = (const float*)d_in[8];
  const float* b1     = (const float*)d_in[9];
  const float* w2     = (const float*)d_in[10];
  const float* b2     = (const float*)d_in[11];

  float* out  = (float*)d_out;        // [B,1] first, then attn [L,NI,B,S,S]
  float* attn = out + BB;

  const size_t ROWS = (size_t)NI * BB * SS;  // 18432
  const int nCW  = NI * DD * KPE;            // 786432
  const int nINW = LL * 3 * DD * DD;         // 1179648
  const int nOUW = LL * DD * DD;             // 393216

  char* base = (char*)d_ws;
  __bf16* peB   = (__bf16*)base; base += ROWS * DD * 2;
  __bf16* qkvB  = (__bf16*)base; base += ROWS * 3 * DD * 2;
  __bf16* vTB   = (__bf16*)base; base += ROWS * DD * 2;
  __bf16* oB    = (__bf16*)base; base += ROWS * DD * 2;
  __bf16* cwB   = (__bf16*)base; base += (size_t)nCW * 2;
  __bf16* inwB  = (__bf16*)base; base += (size_t)nINW * 2;
  __bf16* outwB = (__bf16*)base; base += (size_t)nOUW * 2;
  float*  feat  = (float*)base;  base += (size_t)BB * NI * DD * 4;
  float*  hbuf  = (float*)base;

  // weights -> bf16 (once per launch; deterministic)
  cvt_f32_bf16_kernel<<<(nCW + 255) / 256, 256, 0, stream>>>(conv_w, cwB, nCW);
  cvt_f32_bf16_kernel<<<(nINW + 255) / 256, 256, 0, stream>>>(inw, inwB, nINW);
  cvt_f32_bf16_kernel<<<(nOUW + 255) / 256, 256, 0, stream>>>(outw, outwB, nOUW);

  patch_embed_kernel<<<dim3(36, 8, NI), 256, 0, stream>>>(x, cwB, conv_b, pos,
                                                          peB);

  for (int l = 0; l < LL; ++l) {
    gemm_bias_kernel<<<dim3(144, 24), 256, 0, stream>>>(
        peB, inwB + (size_t)l * 3 * DD * DD, inb + (size_t)l * 3 * DD, qkvB,
        vTB, DD, 3 * DD);
    attention_kernel<<<dim3(36, 32), 256, 0, stream>>>(
        qkvB, vTB, oB, attn + (size_t)l * NI * BB * SS * SS);
    gemm_bias_kernel<<<dim3(144, 8), 256, 0, stream>>>(
        oB, outwB + (size_t)l * DD * DD, outb + (size_t)l * DD, peB, nullptr,
        DD, DD);
  }

  feat_mean_kernel<<<32, 256, 0, stream>>>(peB, feat);
  head1_kernel<<<16, 256, 0, stream>>>(feat, w1, b1, hbuf);
  head2_kernel<<<1, 32, 0, stream>>>(hbuf, w2, b2, out);
}
